// Net_82145544503447
// MI455X (gfx1250) — 8-bit matrix operand order repaired and checked by reading, compiled, not run on hardware
//
#include <hip/hip_runtime.h>


typedef int v8i __attribute__((ext_vector_type(8)));
typedef unsigned int u32x4 __attribute__((ext_vector_type(4)));
typedef unsigned int u32x8 __attribute__((ext_vector_type(8)));

typedef __attribute__((address_space(3))) _Float16 lds_f16;

#define NT    256
#define CH    64
#define R0    24
#define R0SQ  576
#define RIN   26
#define R3    18

#define OFF_X      0
#define OFF_WIN    2704
#define OFF_WOUT   5008
#define OFF_BIAS   7312
#define OFF_BNS    9104
#define OFF_BNB    10896
#define OFF_BOUT   12688
#define OFF_YSUM   12704
#define OFF_SA     54176
#define OFF_SB     91040
#define OFF_WD     127904
#define OFF_WP     129632
#define LDS_TOTAL  141920

__device__ __forceinline__ float ht(float v) {
    return fminf(1.0f, fmaxf(-1.0f, v));
}

__device__ __forceinline__ void tdm_load_f32(const void* gptr,
                                             unsigned lds_off, unsigned n)
{
    unsigned long long ga = (unsigned long long)gptr;
    u32x4 g0;
    g0.x = 1u;
    g0.y = lds_off;
    g0.z = (unsigned)(ga & 0xffffffffull);
    g0.w = (unsigned)((ga >> 32) & 0x1ffffffull)
         | 0x80000000u;
    u32x8 g1;
    g1.s0 = 2u << 16;
    g1.s1 = (n & 0xffffu) << 16;
    g1.s2 = ((n >> 16) & 0xffffu) | (1u << 16);
    g1.s3 = (n & 0xffffu) << 16;
    g1.s4 = 1u;
    g1.s5 = n;
    g1.s6 = 0u;
    g1.s7 = 0u;
    asm volatile("tensor_load_to_lds %0, %1" :: "s"(g0), "s"(g1) : "memory");
}

__device__ __forceinline__ void dw_stage(int RING,
        const signed char* __restrict__ src, signed char* __restrict__ dst,
        const signed char* __restrict__ wd9,
        const float* __restrict__ bias, const float* __restrict__ bns,
        const float* __restrict__ bnb, int gy0, int gx0, int tid)
{
    for (int idx = tid; idx < R0SQ * CH; idx += NT) {
        int pix = idx >> 6, ch = idx & 63;
        int py = pix / R0, px = pix - py * R0;
        const signed char* wr = wd9 + ch * 9;
        int s = 0;
#pragma unroll
        for (int ky = 0; ky < 3; ky++)
#pragma unroll
            for (int kx = 0; kx < 3; kx++)
                s += (int)src[((py - 1 + ky) * R0 + (px - 1 + kx)) * CH + ch] *
                     (int)wr[ky * 3 + kx];
        float v = (float)s + bias[ch];
        v = ht(v * bns[ch] + bnb[ch]);
        bool ring = (py >= RING) && (py < R0 - RING) &&
                    (px >= RING) && (px < R0 - RING);
        bool img  = ((unsigned)(gy0 + py) < 256u) &&
                    ((unsigned)(gx0 + px) < 256u);
        signed char sv = (v >= 0.0f) ? 1 : -1;
        dst[idx] = (ring && img) ? sv : (signed char)0;
    }
}

__device__ __forceinline__ void pw_stage(int RING,
        const signed char* __restrict__ src, signed char* __restrict__ dst,
        lds_f16* ysum, const signed char* __restrict__ wp,
        const float* __restrict__ bias, const float* __restrict__ bns,
        const float* __restrict__ bnb, int gy0, int gx0, int tid)
{
    const int lane = tid & 31;
    const int wid  = tid >> 5;
    const int n    = lane & 15;
    const int half = lane >> 4;

    for (int t = 0; t < 4; ++t) {
        v8i B;
        const signed char* bp = wp + (t * 16 + n) * CH + half * 16;
#pragma unroll
        for (int v = 0; v < 4; v++) {
            B[v]     = *(const int*)(bp + v * 4);
            B[v + 4] = *(const int*)(bp + 32 + v * 4);
        }
        int   ch = t * 16 + n;
        float bi = bias[ch], sc = bns[ch], sh = bnb[ch];

        for (int g = wid; g < 36; g += 8) {
            v8i A;
            const signed char* ap = src + (g * 16 + n) * CH;
#pragma unroll
            for (int j = 0; j < 4; j++) {
                unsigned long long q =
                    *(const unsigned long long*)(ap + (j >> 1) * 32 + half * 16 + (j & 1) * 8);
                A[2 * j]     = (int)(unsigned int)(q & 0xffffffffull);
                A[2 * j + 1] = (int)(unsigned int)(q >> 32);
            }
            v8i Cz = {0, 0, 0, 0, 0, 0, 0, 0};
            v8i D = __builtin_amdgcn_wmma_i32_16x16x64_iu8(
                         true, A,  true, B, Cz,
                         false,  false);

            bool okw[8];
            bool okr[8];
            bool oki[8];
            int  adr[8];
#pragma unroll
            for (int r = 0; r < 8; r++) {
                int p  = g * 16 + half * 8 + r;
                int py = p / R0, px = p - py * R0;
                bool ring = (py >= RING) && (py < R0 - RING) &&
                            (px >= RING) && (px < R0 - RING);
                bool img  = ((unsigned)(gy0 + py) < 256u) &&
                            ((unsigned)(gx0 + px) < 256u);
                okw[r] = ring && img;
                okr[r] = ring;
                oki[r] = img;
                adr[r] = dst ? (p * CH + ch)
                             : ((py - RING) * R3 + (px - RING)) * CH + ch;
            }
#pragma unroll
            for (int r = 0; r < 8; r++) {
                float v = (float)D[r] + bi;
                v = ht(v * sc + sh);
                if (dst) {
                    signed char sv = (v >= 0.0f) ? 1 : -1;
                    dst[adr[r]] = okw[r] ? sv : (signed char)0;
                } else if (okr[r]) {
                    lds_f16* h = ysum + adr[r];
                    float cur = (float)*h;
                    *h = (_Float16)(cur + (oki[r] ? v : 0.0f));
                }
            }
        }
    }
}

extern "C" __global__ __launch_bounds__(NT)
void fused_bnn_kernel(const float* __restrict__ x,
                      const float* __restrict__ w_in,  const float* __restrict__ b_in,
                      const float* __restrict__ w_d1,  const float* __restrict__ b_d1,
                      const float* __restrict__ w_p1,  const float* __restrict__ b_p1,
                      const float* __restrict__ w_d2,  const float* __restrict__ b_d2,
                      const float* __restrict__ w_p2,  const float* __restrict__ b_p2,
                      const float* __restrict__ w_d3,  const float* __restrict__ b_d3,
                      const float* __restrict__ w_p3,  const float* __restrict__ b_p3,
                      const float* __restrict__ w_out, const float* __restrict__ b_out,
                      const float* __restrict__ bn_g,  const float* __restrict__ bn_b,
                      const float* __restrict__ bn_m,  const float* __restrict__ bn_v,
                      float* __restrict__ out)
{
    extern __shared__ char smem[];
    float*       s_x    = (float*)(smem + OFF_X);
    float*       s_win  = (float*)(smem + OFF_WIN);
    float*       s_wout = (float*)(smem + OFF_WOUT);
    float*       s_bias = (float*)(smem + OFF_BIAS);
    float*       s_bns  = (float*)(smem + OFF_BNS);
    float*       s_bnb  = (float*)(smem + OFF_BNB);
    float*       s_bout = (float*)(smem + OFF_BOUT);
    lds_f16*     s_ysum = (lds_f16*)(_Float16*)(smem + OFF_YSUM);
    signed char* s_sa   = (signed char*)(smem + OFF_SA);
    signed char* s_sb   = (signed char*)(smem + OFF_SB);
    signed char* s_wd   = (signed char*)(smem + OFF_WD);
    signed char* s_wp   = (signed char*)(smem + OFF_WP);

    const int tid  = threadIdx.x;
    const int tx   = blockIdx.x, ty = blockIdx.y, bimg = blockIdx.z;
    const int gy0  = ty * 16 - 4;
    const int gx0  = tx * 16 - 4;
    const float* xin = x + bimg * 65536;

    if (tid < 32) {
        unsigned lds0 = (unsigned)(unsigned long long)smem + OFF_SA;
        tdm_load_f32(w_p1, lds0,              4096);
        tdm_load_f32(w_p2, lds0 + 16384u,     4096);
        tdm_load_f32(w_p3, lds0 + 32768u,     4096);
        __builtin_amdgcn_s_wait_tensorcnt(0);
    }

    for (int i = tid; i < RIN * RIN; i += NT) {
        int r = i / RIN, c = i - r * RIN;
        int gy = ty * 16 - 5 + r, gx = tx * 16 - 5 + c;
        bool img = ((unsigned)gy < 256u) && ((unsigned)gx < 256u);
        if (img) __builtin_prefetch(xin + gy * 256 + gx, 0, 1);
        s_x[i] = img ? xin[gy * 256 + gx] : 0.0f;
    }
    for (int i = tid; i < 576; i += NT) { s_win[i] = w_in[i]; s_wout[i] = w_out[i]; }
    {
        const float* bs[7] = {b_in, b_d1, b_p1, b_d2, b_p2, b_d3, b_p3};
        for (int i = tid; i < 7 * CH; i += NT) {
            s_bias[i] = bs[i >> 6][i & 63];
            float sc = bn_g[i] * rsqrtf(bn_v[i] + 1e-5f);
            s_bns[i] = sc;
            s_bnb[i] = bn_b[i] - bn_m[i] * sc;
        }
    }
    {
        const float* wd[3] = {w_d1, w_d2, w_d3};
        for (int i = tid; i < 3 * 576; i += NT)
            s_wd[i] = (wd[i / 576][i % 576] >= 0.0f) ? 1 : -1;
    }
    if (tid == 0) *s_bout = b_out[0];
    __syncthreads();

    {
        const float* raw = (const float*)(smem + OFF_SA);
        for (int i = tid; i < 3 * 4096; i += NT)
            s_wp[i] = (raw[i] >= 0.0f) ? 1 : -1;
    }
    __syncthreads();

    for (int idx = tid; idx < R0SQ * CH; idx += NT) {
        int pix = idx >> 6, ch = idx & 63;
        int py = pix / R0, px = pix - py * R0;
        const float* wr = s_win + ch * 9;
        float v = s_bias[ch];
#pragma unroll
        for (int ky = 0; ky < 3; ky++)
#pragma unroll
            for (int kx = 0; kx < 3; kx++)
                v += s_x[(py + ky) * RIN + px + kx] * wr[ky * 3 + kx];
        v = ht(v * s_bns[ch] + s_bnb[ch]);
        bool img = ((unsigned)(gy0 + py) < 256u) && ((unsigned)(gx0 + px) < 256u);
        s_sa[idx] = img ? (signed char)(v >= 0.0f ? 1 : -1) : (signed char)0;
        if (py >= 3 && py < 21 && px >= 3 && px < 21)
            s_ysum[((py - 3) * R3 + (px - 3)) * CH + ch] =
                (_Float16)(img ? v : 0.0f);
    }
    __syncthreads();

    dw_stage(1, s_sa, s_sb, s_wd + 0 * 576, s_bias + 1 * CH,
             s_bns + 1 * CH, s_bnb + 1 * CH, gy0, gx0, tid);
    __syncthreads();
    pw_stage(1, s_sb, s_sa, nullptr, s_wp + 0 * 4096, s_bias + 2 * CH,
             s_bns + 2 * CH, s_bnb + 2 * CH, gy0, gx0, tid);
    __syncthreads();
    dw_stage(2, s_sa, s_sb, s_wd + 1 * 576, s_bias + 3 * CH,
             s_bns + 3 * CH, s_bnb + 3 * CH, gy0, gx0, tid);
    __syncthreads();
    pw_stage(2, s_sb, s_sa, nullptr, s_wp + 1 * 4096, s_bias + 4 * CH,
             s_bns + 4 * CH, s_bnb + 4 * CH, gy0, gx0, tid);
    __syncthreads();
    dw_stage(3, s_sa, s_sb, s_wd + 2 * 576, s_bias + 5 * CH,
             s_bns + 5 * CH, s_bnb + 5 * CH, gy0, gx0, tid);
    __syncthreads();
    pw_stage(3, s_sb, nullptr, s_ysum, s_wp + 2 * 4096, s_bias + 6 * CH,
             s_bns + 6 * CH, s_bnb + 6 * CH, gy0, gx0, tid);
    __syncthreads();

    {
        int oy = tid >> 4, ox = tid & 15;
        float acc = *s_bout;
        for (int ch = 0; ch < CH; ch++) {
            const float* wr = s_wout + ch * 9;
#pragma unroll
            for (int ky = 0; ky < 3; ky++)
#pragma unroll
                for (int kx = 0; kx < 3; kx++)
                    acc += (float)s_ysum[((oy + ky) * R3 + ox + kx) * CH + ch] *
                           wr[ky * 3 + kx];
        }
        acc += s_x[(oy + 5) * RIN + ox + 5];
        out[bimg * 65536 + (ty * 16 + oy) * 256 + tx * 16 + ox] = acc;
    }
}

extern "C" void kernel_launch(void* const* d_in, const int* in_sizes, int n_in,
                              void* d_out, int out_size, void* d_ws, size_t ws_size,
                              hipStream_t stream) {
    (void)in_sizes; (void)n_in; (void)out_size; (void)d_ws; (void)ws_size;
    const float* x     = (const float*)d_in[0];
    const float* w_in  = (const float*)d_in[1];
    const float* b_in  = (const float*)d_in[2];
    const float* w_d1  = (const float*)d_in[3];
    const float* b_d1  = (const float*)d_in[4];
    const float* w_p1  = (const float*)d_in[5];
    const float* b_p1  = (const float*)d_in[6];
    const float* w_d2  = (const float*)d_in[7];
    const float* b_d2  = (const float*)d_in[8];
    const float* w_p2  = (const float*)d_in[9];
    const float* b_p2  = (const float*)d_in[10];
    const float* w_d3  = (const float*)d_in[11];
    const float* b_d3  = (const float*)d_in[12];
    const float* w_p3  = (const float*)d_in[13];
    const float* b_p3  = (const float*)d_in[14];
    const float* w_out = (const float*)d_in[15];
    const float* b_out = (const float*)d_in[16];
    const float* bn_g  = (const float*)d_in[17];
    const float* bn_b  = (const float*)d_in[18];
    const float* bn_m  = (const float*)d_in[19];
    const float* bn_v  = (const float*)d_in[20];

    dim3 grid(16, 16, 8);
    dim3 block(NT);
    fused_bnn_kernel<<<grid, block, LDS_TOTAL, stream>>>(
        x, w_in, b_in, w_d1, b_d1, w_p1, b_p1, w_d2, b_d2, w_p2, b_p2,
        w_d3, b_d3, w_p3, b_p3, w_out, b_out, bn_g, bn_b, bn_m, bn_v,
        (float*)d_out);
}
